// MiniUNetV1_52682068853171
// MI455X (gfx1250) — matrix operand order repaired and checked by reading, not compiled, not run on hardware
//
#include <hip/hip_runtime.h>
#include <cstdint>
#include <cstddef>

// ---------------------------------------------------------------------------
// MiniUNet (dense spconv emulation) for MI455X / gfx1250.
// All convs are implicit GEMM on v_wmma_f32_16x16x32_bf16 (wave32).
// Compute-bound (~75 GFLOP vs ~0.5 GB HBM) -> feed the matrix pipe:
//   - templated <KD,KH,KW,CIN,NT>: tap loops unroll, all divisors constant
//   - CIN%32==0: tap-outer loop, per-tap validity hoisted; inner loop is
//     2x b128 (A) + 2*NT x b128 (B) + NT x wmma + pointer bumps only
//   - CIN==16: chunk loop unrolls, tap = 2c+gs is compile-time constant
//   - boundary handling by address-clamp + cndmask (no exec-mask branches)
//   - global_prefetch_b8 on the packed-B stream
// ---------------------------------------------------------------------------

typedef __bf16 bf16_t;
typedef bf16_t   v16bf __attribute__((ext_vector_type(16)));
typedef float    v8f_t __attribute__((ext_vector_type(8)));
typedef unsigned uv4   __attribute__((ext_vector_type(4)));

__device__ __forceinline__ unsigned short f2bf_bits(float f) {
  union { float f; unsigned u; } v; v.f = f;
  unsigned r = v.u + 0x7FFFu + ((v.u >> 16) & 1u);   // round-to-nearest-even
  return (unsigned short)(r >> 16);
}
__device__ __forceinline__ bf16_t bits2bf(unsigned short h) {
  return __builtin_bit_cast(bf16_t, h);
}

// ---------------------------------------------------------------------------
// Scatter voxels -> dense bf16 grid + fp32 occupancy mask
// ---------------------------------------------------------------------------
__global__ void scatter_voxels(const float* __restrict__ feats,
                               const int* __restrict__ coords,
                               unsigned short* __restrict__ x0,
                               float* __restrict__ m0,
                               int nv, int H, int W, int Cin) {
  int i = blockIdx.x * blockDim.x + threadIdx.x;
  if (i >= nv) return;
  int z = coords[i * 4 + 1];
  int y = coords[i * 4 + 2];
  int x = coords[i * 4 + 3];
  size_t s = ((size_t)z * H + y) * W + x;
  for (int c = 0; c < Cin; ++c)
    x0[s * Cin + c] = f2bf_bits(feats[(size_t)i * Cin + c]);
  m0[s] = 1.0f;
}

// ---------------------------------------------------------------------------
// Mask dilation: out = (conv(mask, ones) > 0), generic 3D (2D via D=1)
// ---------------------------------------------------------------------------
__global__ void dilate_mask(const float* __restrict__ mi, float* __restrict__ mo,
                            int D, int H, int W,
                            int Do, int Ho, int Wo,
                            int KD, int KH, int KW,
                            int sd, int sh, int sw,
                            int pd, int ph, int pw) {
  int idx = blockIdx.x * blockDim.x + threadIdx.x;
  int M = Do * Ho * Wo;
  if (idx >= M) return;
  int od = idx / (Ho * Wo);
  int r  = idx - od * (Ho * Wo);
  int oy = r / Wo;
  int ox = r - oy * Wo;
  float v = 0.f;
  for (int kz = 0; kz < KD; ++kz)
    for (int ky = 0; ky < KH; ++ky)
      for (int kx = 0; kx < KW; ++kx) {
        int iz = od * sd - pd + kz;
        int iy = oy * sh - ph + ky;
        int ix = ox * sw - pw + kx;
        if (iz >= 0 && iz < D && iy >= 0 && iy < H && ix >= 0 && ix < W)
          v += mi[((size_t)iz * H + iy) * W + ix];
      }
  mo[idx] = (v > 0.f) ? 1.f : 0.f;
}

// max over z (BEV mask)
__global__ void maxz_mask(const float* __restrict__ mi, float* __restrict__ mo,
                          int D, int HW) {
  int i = blockIdx.x * blockDim.x + threadIdx.x;
  if (i >= HW) return;
  float v = 0.f;
  for (int d = 0; d < D; ++d) v = fmaxf(v, mi[(size_t)d * HW + i]);
  mo[i] = v;
}

// (Dz, HW, C) -> (HW, C*Dz), channel = c*Dz + d  (torch view(N, C*D, H, W))
__global__ void bev_reshape(const unsigned short* __restrict__ h,
                            unsigned short* __restrict__ xb,
                            int Dz, int HW, int C) {
  int i = blockIdx.x * blockDim.x + threadIdx.x;
  int total = Dz * HW * C;
  if (i >= total) return;
  int d = i / (HW * C);
  int r = i - d * (HW * C);
  int s = r / C;
  int c = r - s * C;
  xb[(size_t)s * (C * Dz) + c * Dz + d] = h[i];
}

// ---------------------------------------------------------------------------
// Pack fp32 DHWIO weights -> bf16 B fragments [Kpad/32][Cout/16][32][16]
// element (kc,nt,lane,j) = w[(kc*32 + (j&7) + (j&8)*2 + (lane>>4)*8) * Cout + nt*16 + (lane&15)]
// ---------------------------------------------------------------------------
__global__ void pack_weights(const float* __restrict__ w,
                             unsigned short* __restrict__ pb,
                             int Ktot, int Cout) {
  int NTl = Cout >> 4;
  int Kpad = (Ktot + 31) & ~31;
  int total = Kpad * Cout;                 // == (Kpad/32)*NTl*512
  int i = blockIdx.x * blockDim.x + threadIdx.x;
  if (i >= total) return;
  int j    = i & 15;
  int lane = (i >> 4) & 31;
  int blk  = i >> 9;
  int nt   = blk % NTl;
  int kc   = blk / NTl;
  int k    = kc * 32 + (j & 7) + ((j & 8) << 1) + ((lane >> 4) << 3);
  int col  = nt * 16 + (lane & 15);
  pb[i] = (k < Ktot) ? f2bf_bits(w[(size_t)k * Cout + col]) : (unsigned short)0;
}

// ---------------------------------------------------------------------------
// Templated conv + BN + ReLU + mask as implicit GEMM on WMMA bf16 (wave32).
// One wave: 16 output sites x (NT*16) output channels, K chunks of 32.
//
// Fragment layouts per CDNA5 ISA 7.12.2:
//   A (16x32 bf16): lane&15 = row; lanes<16 K={0..7,16..23}, lanes>=16 {8..15,24..31}
//                   -> each v16bf half = 8 consecutive K = one aligned b128 load
//   B (32x16 bf16): lane&15 = col; lanes<16 K={0..7,16..23}, lanes>=16 {8..15,24..31} (pre-packed)
//   C/D (16x16 f32): VGPR i -> row m0+i (+8 for hi half-wave), N = lane&15
// ---------------------------------------------------------------------------
template <int KD, int KH, int KW, int CIN, int NT>
__global__ __launch_bounds__(32) void conv_wmma_t(
    const unsigned short* __restrict__ in,   // bf16 activations [D*H*W, CIN]
    const unsigned short* __restrict__ pb,   // packed bf16 B fragments
    const float* __restrict__ gam, const float* __restrict__ bet,
    const float* __restrict__ mu,  const float* __restrict__ var,
    const float* __restrict__ mask,          // fp32 per output site
    unsigned short* __restrict__ out_bf,     // bf16 output (or null)
    float* __restrict__ out_f32,             // fp32 output (or null)
    int D, int H, int W,
    int Do, int Ho, int Wo,
    int sd, int sh, int sw,
    int pd, int ph, int pw,
    int ldd, int lhh, int lww,               // lhs (input) dilation, 1 = none
    int CoutTotal, int Coff)
{
  constexpr int KHW  = KH * KW;
  constexpr int KVOL = KD * KH * KW;
  constexpr int KTOT = KVOL * CIN;
  constexpr int NCHUNK = (KTOT + 31) / 32;
  constexpr int BSTRIDE = NT * 512;          // packed-B elements per K-chunk

  const int lane = threadIdx.x;
  const int hi = lane >> 4;
  const int lo = lane & 15;
  const int m0 = blockIdx.x << 4;
  const int M  = Do * Ho * Wo;

  const int row = m0 + lo;
  const bool rowValid = row < M;
  const int rr = rowValid ? row : 0;
  const int od = rr / (Ho * Wo);
  const int r2 = rr - od * (Ho * Wo);
  const int oy = r2 / Wo;
  const int ox = r2 - oy * Wo;
  const int bz = od * sd - pd;
  const int by = oy * sh - ph;
  const int bx = ox * sw - pw;

  v8f_t acc[NT];
#pragma unroll
  for (int t = 0; t < NT; ++t)
    acc[t] = v8f_t{0.f, 0.f, 0.f, 0.f, 0.f, 0.f, 0.f, 0.f};

  const unsigned short* pbl = pb + (size_t)lane * 16;

  // helper: B loads + NT wmma for one chunk, then bump pbl
  auto do_wmma = [&](const v16bf& Av) {
    __builtin_prefetch(pbl + BSTRIDE, 0, 1);   // global_prefetch_b8 next B chunk
#pragma unroll
    for (int nt = 0; nt < NT; ++nt) {
      union BU { uv4 q[2]; v16bf v; } Bu;
      Bu.q[0] = *(const uv4*)(pbl + nt * 512);
      Bu.q[1] = *(const uv4*)(pbl + nt * 512 + 8);
      acc[nt] = __builtin_amdgcn_wmma_f32_16x16x32_bf16(
          false, Av, false, Bu.v, (short)0, acc[nt], false, false);
    }
    pbl += BSTRIDE;
  };

  if constexpr ((CIN & 31) == 0) {
    // ---- tap-outer path: one chunk never crosses a tap ----
#pragma unroll
    for (int kz = 0; kz < KD; ++kz)
#pragma unroll
      for (int ky = 0; ky < KH; ++ky)
#pragma unroll
        for (int kx = 0; kx < KW; ++kx) {
          const int tz = bz + kz, ty = by + ky, tx = bx + kx;
          int iz = tz, iy = ty, ix = tx;
          bool ok = rowValid && (tz >= 0) && (ty >= 0) && (tx >= 0);
          if (ldd > 1) { ok = ok && (tz % ldd) == 0; iz = tz / ldd; }
          if (lhh > 1) { ok = ok && (ty % lhh) == 0; iy = ty / lhh; }
          if (lww > 1) { ok = ok && (tx % lww) == 0; ix = tx / lww; }
          ok = ok && (iz < D) && (iy < H) && (ix < W);
          // clamp to index 0 instead of branching; zero via select below
          const size_t sidx = ok ? ((((size_t)iz * H + iy) * W + ix) * CIN) : 0;
          const unsigned short* ap = in + sidx + (hi << 3);
          for (int cb = 0; cb < CIN; cb += 32) {
            uv4 q0 = *(const uv4*)(ap + cb);
            uv4 q1 = *(const uv4*)(ap + cb + 16);
            q0 = ok ? q0 : (uv4)0u;
            q1 = ok ? q1 : (uv4)0u;
            union AU { uv4 q[2]; v16bf v; } Au;
            Au.q[0] = q0; Au.q[1] = q1;
            do_wmma(Au.v);
          }
        }
  } else if constexpr (CIN == 16) {
    // ---- chunk covers exactly two taps; tap index folds after unroll ----
#pragma unroll
    for (int c = 0; c < NCHUNK; ++c) {
      union AU { uv4 q[2]; v16bf v; } Au;
#pragma unroll
      for (int gs = 0; gs < 2; ++gs) {
        const int tap = 2 * c + gs;          // compile-time constant after unroll
        uv4 q = (uv4)0u;
        if (tap < KVOL) {
          const int kz = tap / KHW;
          const int t2 = tap - kz * KHW;
          const int ky = t2 / KW;
          const int kx = t2 - ky * KW;
          const int tz = bz + kz, ty = by + ky, tx = bx + kx;
          const bool ok = rowValid && tz >= 0 && ty >= 0 && tx >= 0 &&
                          tz < D && ty < H && tx < W;
          const size_t sidx = ok ? ((((size_t)tz * H + ty) * W + tx) * CIN) : 0;
          q = *(const uv4*)(in + sidx + (hi << 3));
          q = ok ? q : (uv4)0u;
        }
        Au.q[gs] = q;
      }
      do_wmma(Au.v);
    }
  } else {
    // ---- CIN==4 input layer: scalar gather, chunks unroll (taps fold) ----
#pragma unroll
    for (int c = 0; c < NCHUNK; ++c) {
      const int kb = c * 32;
      union AU { uv4 q[2]; v16bf v; } Au;
#pragma unroll
      for (int j = 0; j < 16; ++j) {
        const int kl = ((j & 8) << 1) + (j & 7) + (hi << 3);
        const int k = kb + kl;
        unsigned short abits = 0;
        if (rowValid && k < KTOT) {
          const int tap = k / CIN;
          const int ci  = k - tap * CIN;
          const int kz  = tap / KHW;
          const int t2  = tap - kz * KHW;
          const int ky  = t2 / KW;
          const int kx  = t2 - ky * KW;
          const int tz = bz + kz, ty = by + ky, tx = bx + kx;
          if (tz >= 0 && ty >= 0 && tx >= 0 && tz < D && ty < H && tx < W)
            abits = in[(((size_t)tz * H + ty) * W + tx) * CIN + ci];
        }
        Au.v[j] = bits2bf(abits);
      }
      do_wmma(Au.v);
    }
  }

  // fused BN + ReLU + mask, store (Cout is always a multiple of 16 here)
#pragma unroll
  for (int nt = 0; nt < NT; ++nt) {
    const int col  = nt * 16 + lo;
    const float s    = gam[col] * rsqrtf(var[col] + 1e-3f);
    const float bias = bet[col];
    const float mean = mu[col];
#pragma unroll
    for (int i = 0; i < 8; ++i) {
      const int orow = m0 + i + (hi << 3);
      if (orow < M) {
        float y = (acc[nt][i] - mean) * s + bias;
        y = fmaxf(y, 0.f);
        y *= mask[orow];
        const size_t oi = (size_t)orow * CoutTotal + Coff + col;
        if (out_f32) out_f32[oi] = y;
        else         out_bf[oi]  = f2bf_bits(y);
      }
    }
  }
}

// ---------------------------------------------------------------------------
// Host side
// ---------------------------------------------------------------------------
namespace {

enum Blk {
  B_CIN = 0, B_C1_0, B_C2_0, B_C2_1, B_C2_2,
  B_C3_0, B_C3_1, B_C3_2, B_C4_0, B_C4_1, B_C4_2, B_C5_0,
  B_M1_0, B_M1_1, B_M1_2, B_M1_3, B_M2_0, B_M2_1, B_M2_2, B_M2_3,
  B_MINV1, B_MINV2, B_MOUT, N_BLK
};

struct Params {
  void* const* d_in;
  const float* w(int b) const { return (const float*)d_in[2 + b * 5 + 0]; }
  const float* g(int b) const { return (const float*)d_in[2 + b * 5 + 1]; }
  const float* b(int b) const { return (const float*)d_in[2 + b * 5 + 2]; }
  const float* m(int b) const { return (const float*)d_in[2 + b * 5 + 3]; }
  const float* v(int b) const { return (const float*)d_in[2 + b * 5 + 4]; }
};

template <int KD, int KH, int KW, int CIN, int NT>
inline void convT(hipStream_t st, const Params& P, int blk,
                  const void* in, const unsigned short* pb, const float* mask,
                  void* out_bf, float* out_f32,
                  int D, int H, int W, int Do, int Ho, int Wo,
                  int sd, int sh, int sw, int pd, int ph, int pw,
                  int ldd = 1, int lhh = 1, int lww = 1,
                  int ctot = -1, int coff = 0) {
  if (ctot < 0) ctot = NT * 16;
  int M = Do * Ho * Wo;
  dim3 grid((unsigned)((M + 15) / 16));
  conv_wmma_t<KD, KH, KW, CIN, NT><<<grid, 32, 0, st>>>(
      (const unsigned short*)in, pb,
      P.g(blk), P.b(blk), P.m(blk), P.v(blk), mask,
      (unsigned short*)out_bf, out_f32,
      D, H, W, Do, Ho, Wo, sd, sh, sw, pd, ph, pw, ldd, lhh, lww, ctot, coff);
}

} // namespace

extern "C" void kernel_launch(void* const* d_in, const int* in_sizes, int n_in,
                              void* d_out, int out_size, void* d_ws, size_t ws_size,
                              hipStream_t stream) {
  (void)in_sizes; (void)n_in; (void)out_size; (void)ws_size;
  Params P{d_in};
  const float* feats  = (const float*)d_in[0];
  const int*   coords = (const int*)d_in[1];
  const int NV = 60000, CIN = 4;

  // level dims (z, y, x)
  const int D0 = 41, H0 = 200, W0 = 200;
  const int D1 = 21, H1 = 100, W1 = 100;
  const int D2 = 11, H2 = 50,  W2 = 50;
  const int D3 = 5,  H3 = 25,  W3 = 25;
  const int D4 = 2,  H4 = 25,  W4 = 25;
  const size_t N0 = (size_t)D0 * H0 * W0;
  const size_t N1 = (size_t)D1 * H1 * W1;
  const size_t N3 = (size_t)D3 * H3 * W3;
  const size_t N4 = (size_t)D4 * H4 * W4;
  const size_t NB = 25 * 25, NBS = 13 * 13;

  // ---- bump allocator over d_ws (~140 MB) ----
  char* base = (char*)d_ws;
  size_t off = 0;
  auto alloc = [&](size_t bytes) -> void* {
    void* p = base + off;
    off = (off + bytes + 255) & ~(size_t)255;
    return p;
  };
  unsigned short* x0 = (unsigned short*)alloc(N0 * CIN * 2);
  float* maskA = (float*)alloc(N0 * 4);          // m0 -> m2 -> m4
  float* maskB = (float*)alloc(N1 * 4);          // m1 -> m3
  unsigned short* Pg0 = (unsigned short*)alloc(N0 * 16 * 2);
  unsigned short* Pg1 = (unsigned short*)alloc(N0 * 16 * 2);
  unsigned short* XB  = (unsigned short*)alloc(NB * 256 * 2);
  unsigned short* S0  = (unsigned short*)alloc(NB * 256 * 2);
  unsigned short* S1  = (unsigned short*)alloc(NB * 256 * 2);
  unsigned short* S2  = (unsigned short*)alloc(NB * 256 * 2);
  unsigned short* CC  = (unsigned short*)alloc(NB * 256 * 2);
  float* mbev = (float*)alloc(NB * 4);
  float* maM  = (float*)alloc(NB * 4);
  float* mbM  = (float*)alloc(NBS * 4);

  // ---- packed weight buffers + prep kernels ----
  static const int KTOTS[N_BLK] = { 108, 432, 432, 864, 864, 864, 1728, 1728,
                                    1728, 1728, 1728, 192,
                                    2304, 1152, 1152, 1152, 1152, 1152, 1152, 1152,
                                    1152, 1152, 2304 };
  static const int COUTS[N_BLK] = { 16, 16, 32, 32, 32, 64, 64, 64, 64, 64, 64, 128,
                                    128, 128, 128, 128, 128, 128, 128, 128,
                                    128, 128, 128 };
  unsigned short* PB[N_BLK];
  for (int b = 0; b < N_BLK; ++b) {
    int kpad = (KTOTS[b] + 31) & ~31;
    size_t total = (size_t)kpad * COUTS[b];
    PB[b] = (unsigned short*)alloc(total * 2);
    pack_weights<<<(unsigned)((total + 255) / 256), 256, 0, stream>>>(
        P.w(b), PB[b], KTOTS[b], COUTS[b]);
  }

  // ---- scatter to dense grid ----
  (void)hipMemsetAsync(x0, 0, N0 * CIN * 2, stream);
  (void)hipMemsetAsync(maskA, 0, N0 * 4, stream);
  scatter_voxels<<<(NV + 255) / 256, 256, 0, stream>>>(feats, coords, x0, maskA,
                                                       NV, H0, W0, CIN);

  // ---- 3D backbone ----
  convT<3,3,3,4,1>(stream, P, B_CIN,  x0,  PB[B_CIN],  maskA, Pg0, nullptr,
                   D0,H0,W0, D0,H0,W0, 1,1,1, 1,1,1);
  convT<3,3,3,16,1>(stream, P, B_C1_0, Pg0, PB[B_C1_0], maskA, Pg1, nullptr,
                   D0,H0,W0, D0,H0,W0, 1,1,1, 1,1,1);

  dilate_mask<<<(D1*H1*W1 + 255) / 256, 256, 0, stream>>>(maskA, maskB,
      D0,H0,W0, D1,H1,W1, 3,3,3, 2,2,2, 1,1,1);
  convT<3,3,3,16,2>(stream, P, B_C2_0, Pg1, PB[B_C2_0], maskB, Pg0, nullptr,
                   D0,H0,W0, D1,H1,W1, 2,2,2, 1,1,1);
  convT<3,3,3,32,2>(stream, P, B_C2_1, Pg0, PB[B_C2_1], maskB, Pg1, nullptr,
                   D1,H1,W1, D1,H1,W1, 1,1,1, 1,1,1);
  convT<3,3,3,32,2>(stream, P, B_C2_2, Pg1, PB[B_C2_2], maskB, Pg0, nullptr,
                   D1,H1,W1, D1,H1,W1, 1,1,1, 1,1,1);

  dilate_mask<<<(D2*H2*W2 + 255) / 256, 256, 0, stream>>>(maskB, maskA,
      D1,H1,W1, D2,H2,W2, 3,3,3, 2,2,2, 1,1,1);
  convT<3,3,3,32,4>(stream, P, B_C3_0, Pg0, PB[B_C3_0], maskA, Pg1, nullptr,
                   D1,H1,W1, D2,H2,W2, 2,2,2, 1,1,1);
  convT<3,3,3,64,4>(stream, P, B_C3_1, Pg1, PB[B_C3_1], maskA, Pg0, nullptr,
                   D2,H2,W2, D2,H2,W2, 1,1,1, 1,1,1);
  convT<3,3,3,64,4>(stream, P, B_C3_2, Pg0, PB[B_C3_2], maskA, Pg1, nullptr,
                   D2,H2,W2, D2,H2,W2, 1,1,1, 1,1,1);

  dilate_mask<<<((int)N3 + 255) / 256, 256, 0, stream>>>(maskA, maskB,
      D2,H2,W2, D3,H3,W3, 3,3,3, 2,2,2, 0,1,1);
  convT<3,3,3,64,4>(stream, P, B_C4_0, Pg1, PB[B_C4_0], maskB, Pg0, nullptr,
                   D2,H2,W2, D3,H3,W3, 2,2,2, 0,1,1);
  convT<3,3,3,64,4>(stream, P, B_C4_1, Pg0, PB[B_C4_1], maskB, Pg1, nullptr,
                   D3,H3,W3, D3,H3,W3, 1,1,1, 1,1,1);
  convT<3,3,3,64,4>(stream, P, B_C4_2, Pg1, PB[B_C4_2], maskB, Pg0, nullptr,
                   D3,H3,W3, D3,H3,W3, 1,1,1, 1,1,1);

  dilate_mask<<<((int)N4 + 255) / 256, 256, 0, stream>>>(maskB, maskA,
      D3,H3,W3, D4,H4,W4, 3,1,1, 2,1,1, 0,0,0);
  convT<3,1,1,64,8>(stream, P, B_C5_0, Pg0, PB[B_C5_0], maskA, Pg1, nullptr,
                   D3,H3,W3, D4,H4,W4, 2,1,1, 0,0,0);

  // ---- 3D -> BEV ----
  maxz_mask<<<((int)NB + 255) / 256, 256, 0, stream>>>(maskA, mbev, D4, (int)NB);
  dilate_mask<<<((int)NB + 255) / 256, 256, 0, stream>>>(mbev, maM,
      1,25,25, 1,25,25, 1,3,3, 1,1,1, 0,1,1);
  {
    int total = D4 * (int)NB * 128;
    bev_reshape<<<(total + 255) / 256, 256, 0, stream>>>(Pg1, XB, D4, (int)NB, 128);
  }

  // ---- 2D middle: block 1 @ 25x25 ----
  convT<1,3,3,256,8>(stream, P, B_M1_0, XB, PB[B_M1_0], maM, S1, nullptr,
                   1,25,25, 1,25,25, 1,1,1, 0,1,1);
  convT<1,3,3,128,8>(stream, P, B_M1_1, S1, PB[B_M1_1], maM, S2, nullptr,
                   1,25,25, 1,25,25, 1,1,1, 0,1,1);
  convT<1,3,3,128,8>(stream, P, B_M1_2, S2, PB[B_M1_2], maM, S1, nullptr,
                   1,25,25, 1,25,25, 1,1,1, 0,1,1);
  convT<1,3,3,128,8>(stream, P, B_M1_3, S1, PB[B_M1_3], maM, S0, nullptr,
                   1,25,25, 1,25,25, 1,1,1, 0,1,1);

  // ---- 2D middle: block 2 @ 13x13 ----
  dilate_mask<<<((int)NBS + 255) / 256, 256, 0, stream>>>(maM, mbM,
      1,25,25, 1,13,13, 1,3,3, 1,2,2, 0,1,1);
  convT<1,3,3,128,8>(stream, P, B_M2_0, S0, PB[B_M2_0], mbM, S1, nullptr,
                   1,25,25, 1,13,13, 1,2,2, 0,1,1);
  convT<1,3,3,128,8>(stream, P, B_M2_1, S1, PB[B_M2_1], mbM, S2, nullptr,
                   1,13,13, 1,13,13, 1,1,1, 0,1,1);
  convT<1,3,3,128,8>(stream, P, B_M2_2, S2, PB[B_M2_2], mbM, S1, nullptr,
                   1,13,13, 1,13,13, 1,1,1, 0,1,1);
  convT<1,3,3,128,8>(stream, P, B_M2_3, S1, PB[B_M2_3], mbM, S2, nullptr,
                   1,13,13, 1,13,13, 1,1,1, 0,1,1);

  // ---- inverse path + concat into CC (256 ch) ----
  convT<1,3,3,128,8>(stream, P, B_MINV1, S0, PB[B_MINV1], maM, CC, nullptr,
                   1,25,25, 1,25,25, 1,1,1, 0,1,1, 1,1,1, /*ctot=*/256, /*coff=*/0);
  // transposed conv: lhs_dilation (2,2), pad 1 -> 13x13 up to 25x25
  convT<1,3,3,128,8>(stream, P, B_MINV2, S2, PB[B_MINV2], maM, CC, nullptr,
                   1,13,13, 1,25,25, 1,1,1, 0,1,1, 1,2,2, /*ctot=*/256, /*coff=*/128);

  // ---- final conv -> fp32 output (1,25,25,128) ----
  convT<1,3,3,256,8>(stream, P, B_MOUT, CC, PB[B_MOUT], maM, nullptr, (float*)d_out,
                   1,25,25, 1,25,25, 1,1,1, 0,1,1);
}
